// PerformerHead_5334349381716
// MI455X (gfx1250) — compile-verified, matrix operand order checked by reading, not run on hardware
//
#include <hip/hip_runtime.h>
#include <hip/hip_bf16.h>
#include <math.h>

// ---------------- problem constants ----------------
#define B_     4
#define S_     2048
#define DIM_   128
#define H_     8
#define DH_    128
#define M_     128
#define INNER_ 1024
#define FF_    512
#define DEPTH_ 2
#define NCLS_  10
#define CHUNK_ 128
#define NC_    16
#define NT_    (B_ * S_)      // 8192 tokens
#define RQ_    (NT_ * H_)     // 65536 (b,s,h) feature rows
#define LDP_   (CHUNK_ + 8)   // padded LDS row stride (halves): 272B rows,
                              // 16B-aligned, 68-DWORD stride -> conflict-free frags

typedef __attribute__((ext_vector_type(16))) _Float16 v16h;
typedef __attribute__((ext_vector_type(8)))  _Float16 v8h;
typedef __attribute__((ext_vector_type(8)))  float    v8f;
typedef _Float16 f16;

// DH^-0.5 == M^-0.5 == 1/sqrt(128)
#define INVSQRT128 0.08838834764831845f
// DH^-0.25
#define NORM_QK    0.29730177875068026f

// ---------------- WMMA helpers ----------------
// Fragment layout (CDNA5 ISA 7.12.2, 16-bit A 16x32):
//   lane<16 : row = lane,    K = {0..7} U {16..23}
//   lane>=16: row = lane-16, K = {8..15} U {24..31}
// B-fragments use the identical indexing when B is stored TRANSPOSED
// (BT[N][K] row-major), so one loader serves both.
__device__ __forceinline__ v16h load_frag(const f16* base, int ld) {
  const int lane = threadIdx.x & 31;
  const int r    = lane & 15;
  const int kb   = (lane < 16) ? 0 : 8;
  const f16* p   = base + (size_t)r * ld + kb;
  v8h lo = *(const v8h*)(p);
  v8h hi = *(const v8h*)(p + 16);
  v16h out;
#pragma unroll
  for (int i = 0; i < 8; ++i) { out[i] = lo[i]; out[i + 8] = hi[i]; }
  return out;
}

__device__ __forceinline__ v8f wmma16(v16h a, v16h b, v8f c) {
  return __builtin_amdgcn_wmma_f32_16x16x32_f16(
      /*neg_a*/false, a, /*neg_b*/false, b,
      /*c_mod*/(short)0, c, /*reuse_a*/false, /*reuse_b*/false);
}

__device__ __forceinline__ float gelu_f(float x) {
  const float k0 = 0.7978845608028654f;   // sqrt(2/pi)
  float x3 = x * x * x;
  return 0.5f * x * (1.0f + tanhf(k0 * (x + 0.044715f * x3)));
}

// ---------------- small elementwise / setup kernels ----------------
__global__ void embed_kernel(const float* __restrict__ x, const float* __restrict__ lin_w,
                             const float* __restrict__ lin_b, const float* __restrict__ pe,
                             float* __restrict__ h) {
  int i = blockIdx.x * 256 + threadIdx.x;          // over NT_*DIM_
  if (i >= NT_ * DIM_) return;
  int d  = i & (DIM_ - 1);
  int bs = i >> 7;
  int s  = bs & (S_ - 1);
  h[i] = x[bs] * lin_w[d] + lin_b[d] + pe[s * DIM_ + d];
}

// cast + transpose: src f32 [R,C] -> dst f16 [C,R]
__global__ void castT_kernel(const float* __restrict__ src, f16* __restrict__ dst,
                             int R, int C) {
  int i = blockIdx.x * 256 + threadIdx.x;
  if (i >= R * C) return;
  int r = i / C, c = i % C;
  dst[(size_t)c * R + r] = (f16)src[i];
}

// projection features, scaled by DH^-0.25; already in BT layout [M,DH]
__global__ void projscale_kernel(const float* __restrict__ proj, f16* __restrict__ dst) {
  int i = blockIdx.x * 256 + threadIdx.x;
  if (i >= M_ * DH_) return;
  dst[i] = (f16)(proj[i] * NORM_QK);
}

// layernorm: one row (DIM_=128) per 128-thread block, f16 output
__global__ void ln_kernel(const float* __restrict__ h, const float* __restrict__ g,
                          const float* __restrict__ bt, f16* __restrict__ y) {
  __shared__ float red[DIM_];
  int row = blockIdx.x, t = threadIdx.x;
  float v = h[(size_t)row * DIM_ + t];
  red[t] = v; __syncthreads();
  for (int s = 64; s > 0; s >>= 1) { if (t < s) red[t] += red[t + s]; __syncthreads(); }
  float mu = red[0] * (1.0f / DIM_);
  __syncthreads();
  float d = v - mu;
  red[t] = d * d; __syncthreads();
  for (int s = 64; s > 0; s >>= 1) { if (t < s) red[t] += red[t + s]; __syncthreads(); }
  float var = red[0] * (1.0f / DIM_);
  float out = d * rsqrtf(var + 1e-5f) * g[t] + bt[t];
  y[(size_t)row * DIM_ + t] = (f16)out;
}

// ---------------- generic WMMA GEMM (32x64 per wave) ----------------
// C[M,N] = A[M,K] @ BT^T ; BT is [N,K] (transposed weights), all f16, f32 acc.
// Block = 8 waves covering 256 rows x 64 cols; each wave owns a 32x64 strip
// (2 m-tiles x 4 n-tiles): A frags reused 4x, B frags reused 2x, and all 8
// waves share the same 64 columns for L0/L2 reuse of BT.
// MODE 0: f16 out               MODE 2: f16 out, +bias, GELU
// MODE 3: f32 out += val + bias (residual accumulate)
template <int MODE>
__global__ __launch_bounds__(256)
void gemm_kernel(const f16* __restrict__ A, const f16* __restrict__ BT,
                 const float* __restrict__ bias, void* __restrict__ Cout,
                 int Mrows, int N, int K) {
  const int wave = threadIdx.x >> 5;
  const int lane = threadIdx.x & 31;
  const int tn0 = blockIdx.x * 4;                 // first of 4 n-tiles
  const int tm0 = blockIdx.y * 16 + wave * 2;     // first of 2 m-tiles
  const f16* A0 = A + (size_t)tm0 * 16 * K;
  const f16* A1 = A0 + (size_t)16 * K;
  const f16* B0 = BT + (size_t)tn0 * 16 * K;

  v8f acc[2][4];
#pragma unroll
  for (int mi = 0; mi < 2; ++mi)
#pragma unroll
    for (int n = 0; n < 4; ++n) { v8f z = {}; acc[mi][n] = z; }

  for (int k = 0; k < K; k += 32) {
    v16h a0 = load_frag(A0 + k, K);
    v16h a1 = load_frag(A1 + k, K);
#pragma unroll
    for (int n = 0; n < 4; ++n) {
      v16h b = load_frag(B0 + (size_t)n * 16 * K + k, K);
      acc[0][n] = wmma16(a0, b, acc[0][n]);
      acc[1][n] = wmma16(a1, b, acc[1][n]);
    }
  }

  const int jloc = lane & 15;
  const int roff = (lane < 16) ? 0 : 8;
#pragma unroll
  for (int n = 0; n < 4; ++n) {
    const int coln = (tn0 + n) * 16 + jloc;
    float bv = (MODE == 0) ? 0.0f : bias[coln];
#pragma unroll
    for (int mi = 0; mi < 2; ++mi) {
#pragma unroll
      for (int j = 0; j < 8; ++j) {
        int row = (tm0 + mi) * 16 + j + roff;
        float v = acc[mi][n][j] + bv;
        if (MODE == 2) v = gelu_f(v);
        size_t idx = (size_t)row * N + coln;
        if (MODE == 3) { ((float*)Cout)[idx] += v; }
        else           { ((f16*)Cout)[idx] = (f16)v; }
      }
    }
  }
}

// ---------------- FAVOR+ feature kernels ----------------
// q path fused: one 128-thread block per (b,s,h) row
__global__ void qfeat_kernel(const f16* __restrict__ q16, const f16* __restrict__ dd16,
                             f16* __restrict__ qp16) {
  __shared__ float red[DIM_];
  int r = blockIdx.x, t = threadIdx.x;
  float qv = (float)q16[(size_t)r * DH_ + t];
  float dd = (float)dd16[(size_t)r * M_ + t];
  red[t] = qv * qv; __syncthreads();
  for (int s = 64; s > 0; s >>= 1) { if (t < s) red[t] += red[t + s]; __syncthreads(); }
  float diag = 0.5f * INVSQRT128 * red[0];      // 0.5*norm^2*sum(q^2)
  __syncthreads();
  red[t] = dd; __syncthreads();
  for (int s = 64; s > 0; s >>= 1) { if (t < s) red[t] = fmaxf(red[t], red[t + s]); __syncthreads(); }
  float stab = red[0];
  qp16[(size_t)r * M_ + t] = (f16)(INVSQRT128 * (expf(dd - diag - stab) + 1e-4f));
}

// k path pass 1: per-row diag + row max
__global__ void krow_kernel(const f16* __restrict__ k16, const f16* __restrict__ dd16,
                            float* __restrict__ diag, float* __restrict__ rowmax) {
  __shared__ float red[DIM_];
  int r = blockIdx.x, t = threadIdx.x;
  float kv = (float)k16[(size_t)r * DH_ + t];
  float dd = (float)dd16[(size_t)r * M_ + t];
  red[t] = kv * kv; __syncthreads();
  for (int s = 64; s > 0; s >>= 1) { if (t < s) red[t] += red[t + s]; __syncthreads(); }
  if (t == 0) diag[r] = 0.5f * INVSQRT128 * red[0];
  __syncthreads();
  red[t] = dd; __syncthreads();
  for (int s = 64; s > 0; s >>= 1) { if (t < s) red[t] = fmaxf(red[t], red[t + s]); __syncthreads(); }
  if (t == 0) rowmax[r] = red[0];
}

// k path pass 2: max over all s for each (b,h)
__global__ void kreduce_kernel(const float* __restrict__ rowmax, float* __restrict__ stabk) {
  __shared__ float red[256];
  int g = blockIdx.x;               // b*H_+hh
  int b = g / H_, hh = g % H_;
  float mx = -1e30f;
  for (int s = threadIdx.x; s < S_; s += 256)
    mx = fmaxf(mx, rowmax[(size_t)(b * S_ + s) * H_ + hh]);
  red[threadIdx.x] = mx; __syncthreads();
  for (int s = 128; s > 0; s >>= 1) {
    if (threadIdx.x < s) red[threadIdx.x] = fmaxf(red[threadIdx.x], red[threadIdx.x + s]);
    __syncthreads();
  }
  if (threadIdx.x == 0) stabk[g] = red[0];
}

// k path pass 3: apply exp
__global__ void kapply_kernel(const f16* __restrict__ dd16, const float* __restrict__ diag,
                              const float* __restrict__ stabk, f16* __restrict__ kp16) {
  size_t i = (size_t)blockIdx.x * 256 + threadIdx.x;   // over RQ_*M_
  if (i >= (size_t)RQ_ * M_) return;
  size_t r  = i >> 7;
  int hh    = (int)(r % H_);
  int b     = (int)((r / H_) >> 11);
  float st  = stabk[b * H_ + hh];
  kp16[i] = (f16)(INVSQRT128 * (expf((float)dd16[i] - diag[r] - st) + 1e-4f));
}

// ---------------- chunked causal linear attention ----------------
// one workgroup (256 thr, 8 waves) per (b,h); 16 sequential chunks of 128.
// kv^T [DH, M] state lives in per-wave f32 WMMA accumulators (wave w owns
// d in [16w,16w+16)); an f16 shadow in LDS feeds the q@kv GEMM.
// LDS tiles use padded stride LDP_ to kill bank conflicts.
__global__ __launch_bounds__(256)
void attn_kernel(const f16* __restrict__ qp, const f16* __restrict__ kp,
                 const f16* __restrict__ vv, f16* __restrict__ oo) {
  __shared__ __align__(16) f16 abuf_s[CHUNK_ * LDP_]; // attn [i][j], later kiT [m][j]
  __shared__ __align__(16) f16 viT_s[CHUNK_ * LDP_];  // v chunk transposed [d][j]
  __shared__ __align__(16) f16 kvT_s[CHUNK_ * LDP_];  // f16 kv^T shadow [d][m]
  __shared__ float ksum_s[CHUNK_];
  __shared__ float den_s[CHUNK_];

  const int g = blockIdx.x;          // b*H_+hh
  const int b = g / H_, hh = g % H_;
  const int tid  = threadIdx.x;
  const int wave = tid >> 5;
  const int lane = tid & 31;
  const int jloc = lane & 15;
  const int roff = (lane < 16) ? 0 : 8;
  const int ldr = H_ * M_;           // row stride (halves) between tokens

  for (int i = tid; i < CHUNK_; i += 256) ksum_s[i] = 0.0f;
  for (int i = tid; i < CHUNK_ * LDP_; i += 256) kvT_s[i] = (f16)0.0f;

  v8f kvacc[8];
#pragma unroll
  for (int t = 0; t < 8; ++t) { v8f z = {}; kvacc[t] = z; }

  __syncthreads();

  for (int c = 0; c < NC_; ++c) {
    const int s0 = c * CHUNK_;
    const size_t tok0 = ((size_t)(b * S_ + s0) * H_ + hh);

    // stage v chunk transposed into LDS
    for (int i = tid; i < CHUNK_ * CHUNK_; i += 256) {
      int j = i >> 7, x = i & 127;
      viT_s[x * LDP_ + j] = vv[(tok0 + (size_t)j * H_) * DH_ + x];
    }
    __syncthreads();

    // 1) attn = tril(qi @ ki^T); wave w -> row tile w, all 8 col tiles
    const int ti = wave;
    for (int tj = 0; tj < 8; ++tj) {
      v8f cf = {};
      if (tj <= ti) {
        const f16* Ab = qp + (tok0 + (size_t)ti * 16 * H_) * M_;
        const f16* Bb = kp + (tok0 + (size_t)tj * 16 * H_) * M_;
        for (int k = 0; k < M_; k += 32)
          cf = wmma16(load_frag(Ab + k, ldr), load_frag(Bb + k, ldr), cf);
        if (tj == ti) {
#pragma unroll
          for (int j = 0; j < 8; ++j) if (j + roff < jloc) cf[j] = 0.0f;
        }
      }
#pragma unroll
      for (int j = 0; j < 8; ++j)
        abuf_s[(ti * 16 + j + roff) * LDP_ + tj * 16 + jloc] = (f16)cf[j];
    }
    __syncthreads();

    // 2) den_i = sum_j attn[i,j] + qi . ksum + 1e-6
    if (tid < CHUNK_) {
      int i = tid;
      float ds = 0.0f;
      for (int j = 0; j < CHUNK_; ++j) ds += (float)abuf_s[i * LDP_ + j];
      const f16* qrow = qp + (tok0 + (size_t)i * H_) * M_;
      float qs = 0.0f;
      for (int m = 0; m < M_; ++m) qs += (float)qrow[m] * ksum_s[m];
      den_s[i] = ds + qs + 1e-6f;
    }
    __syncthreads();

    // 3) out = (attn@vi + qi@kv_sum) / den ; wave w -> row tile w, 8 d-tiles
    for (int td = 0; td < 8; ++td) {
      v8f cf = {};
      const f16* A1 = abuf_s + (ti * 16) * LDP_;
      const f16* B1 = viT_s + (td * 16) * LDP_;
      for (int k = 0; k < CHUNK_; k += 32)
        cf = wmma16(load_frag(A1 + k, LDP_), load_frag(B1 + k, LDP_), cf);
      const f16* A2 = qp + (tok0 + (size_t)ti * 16 * H_) * M_;
      const f16* B2 = kvT_s + (td * 16) * LDP_;
      for (int k = 0; k < M_; k += 32)
        cf = wmma16(load_frag(A2 + k, ldr), load_frag(B2 + k, LDP_), cf);
#pragma unroll
      for (int j = 0; j < 8; ++j) {
        int iloc = j + roff;
        float val = cf[j] / den_s[ti * 16 + iloc];
        oo[(tok0 + (size_t)(ti * 16 + iloc) * H_) * DH_ + td * 16 + jloc] = (f16)val;
      }
    }
    __syncthreads();

    // stage k chunk transposed into abuf (attn no longer needed)
    for (int i = tid; i < CHUNK_ * CHUNK_; i += 256) {
      int j = i >> 7, x = i & 127;
      abuf_s[x * LDP_ + j] = kp[(tok0 + (size_t)j * H_) * M_ + x];
    }
    __syncthreads();

    // 4) kv^T += vi^T @ ki : wave w owns d-tile w, loops 8 m-tiles
    for (int tm = 0; tm < 8; ++tm) {
      v8f cf = kvacc[tm];
      const f16* A = viT_s + (wave * 16) * LDP_;
      const f16* Bt = abuf_s + (tm * 16) * LDP_;   // kiT rows m
      for (int k = 0; k < CHUNK_; k += 32)
        cf = wmma16(load_frag(A + k, LDP_), load_frag(Bt + k, LDP_), cf);
      kvacc[tm] = cf;
#pragma unroll
      for (int j = 0; j < 8; ++j)
        kvT_s[(wave * 16 + j + roff) * LDP_ + tm * 16 + jloc] = (f16)cf[j];
    }
    // 5) ksum[m] += sum_j ki[j,m]
    if (tid < CHUNK_) {
      float s = 0.0f;
      for (int j = 0; j < CHUNK_; ++j) s += (float)abuf_s[tid * LDP_ + j];
      ksum_s[tid] += s;
    }
    __syncthreads();
  }
}

// ---------------- pooling + classifier ----------------
__global__ void pool_kernel(const float* __restrict__ h, float* __restrict__ pooled) {
  int i = blockIdx.x * 256 + threadIdx.x;     // over B_*DIM_
  if (i >= B_ * DIM_) return;
  int b = i >> 7, d = i & 127;
  float s = 0.0f;
  for (int t = 0; t < S_; ++t) s += h[((size_t)b * S_ + t) * DIM_ + d];
  pooled[i] = s * (1.0f / S_);
}

__global__ void final_kernel(const float* __restrict__ pooled, const float* __restrict__ fw,
                             const float* __restrict__ fb, float* __restrict__ out) {
  int i = threadIdx.x;
  if (i >= B_ * NCLS_) return;
  int b = i / NCLS_, c = i % NCLS_;
  float s = fb[c];
  for (int d = 0; d < DIM_; ++d) s += pooled[b * DIM_ + d] * fw[d * NCLS_ + c];
  out[i] = s;
}

// ---------------- host orchestration ----------------
extern "C" void kernel_launch(void* const* d_in, const int* in_sizes, int n_in,
                              void* d_out, int out_size, void* d_ws, size_t ws_size,
                              hipStream_t stream) {
  const float* x     = (const float*)d_in[0];
  const float* lin_w = (const float*)d_in[1];
  const float* lin_b = (const float*)d_in[2];
  const float* pe    = (const float*)d_in[3];
  const float* proj  = (const float*)d_in[4];
  const float* ln1_g = (const float*)d_in[5];
  const float* ln1_b = (const float*)d_in[6];
  const float* wq    = (const float*)d_in[7];
  const float* wk    = (const float*)d_in[8];
  const float* wv    = (const float*)d_in[9];
  const float* wo    = (const float*)d_in[10];
  const float* wo_b  = (const float*)d_in[11];
  const float* ln2_g = (const float*)d_in[12];
  const float* ln2_b = (const float*)d_in[13];
  const float* ff1_w = (const float*)d_in[14];
  const float* ff1_b = (const float*)d_in[15];
  const float* ff2_w = (const float*)d_in[16];
  const float* ff2_b = (const float*)d_in[17];
  const float* fin_w = (const float*)d_in[18];
  const float* fin_b = (const float*)d_in[19];
  float* out = (float*)d_out;

  // workspace carve-up
  char* w = (char*)d_ws;
  size_t off = 0;
  auto alloc = [&](size_t bytes) -> void* {
    void* p = w + off;
    off += (bytes + 255) & ~(size_t)255;
    return p;
  };
  float* h    = (float*)alloc((size_t)NT_ * DIM_ * 4);
  f16* y16    = (f16*)alloc((size_t)NT_ * DIM_ * 2);
  f16* q16    = (f16*)alloc((size_t)NT_ * INNER_ * 2);
  f16* k16    = (f16*)alloc((size_t)NT_ * INNER_ * 2);
  f16* v16    = (f16*)alloc((size_t)NT_ * INNER_ * 2);
  f16* dd16   = (f16*)alloc((size_t)RQ_ * M_ * 2);
  f16* qp16   = (f16*)alloc((size_t)RQ_ * M_ * 2);
  f16* kp16   = (f16*)alloc((size_t)RQ_ * M_ * 2);
  float* diag   = (float*)alloc((size_t)RQ_ * 4);
  float* rowmax = (float*)alloc((size_t)RQ_ * 4);
  float* stabk  = (float*)alloc((size_t)B_ * H_ * 4);
  float* pooled = (float*)alloc((size_t)B_ * DIM_ * 4);
  f16* wqT = (f16*)alloc((size_t)DEPTH_ * INNER_ * DIM_ * 2);
  f16* wkT = (f16*)alloc((size_t)DEPTH_ * INNER_ * DIM_ * 2);
  f16* wvT = (f16*)alloc((size_t)DEPTH_ * INNER_ * DIM_ * 2);
  f16* woT = (f16*)alloc((size_t)DEPTH_ * DIM_ * INNER_ * 2);
  f16* f1T = (f16*)alloc((size_t)DEPTH_ * FF_ * DIM_ * 2);
  f16* f2T = (f16*)alloc((size_t)DEPTH_ * DIM_ * FF_ * 2);
  f16* pjs = (f16*)alloc((size_t)DEPTH_ * M_ * DH_ * 2);
  f16* o16  = q16;   // reuse: q16 dead once qp16 is built
  f16* ff16 = dd16;  // reuse: dd16 dead once kp16 is built

  auto cdiv = [](int a, int b) { return (a + b - 1) / b; };

  // embed: h = x@lin_w + lin_b + pe
  embed_kernel<<<cdiv(NT_ * DIM_, 256), 256, 0, stream>>>(x, lin_w, lin_b, pe, h);

  // weight prep (transposed f16 copies)
  for (int l = 0; l < DEPTH_; ++l) {
    castT_kernel<<<cdiv(DIM_ * INNER_, 256), 256, 0, stream>>>(
        wq + (size_t)l * DIM_ * INNER_, wqT + (size_t)l * INNER_ * DIM_, DIM_, INNER_);
    castT_kernel<<<cdiv(DIM_ * INNER_, 256), 256, 0, stream>>>(
        wk + (size_t)l * DIM_ * INNER_, wkT + (size_t)l * INNER_ * DIM_, DIM_, INNER_);
    castT_kernel<<<cdiv(DIM_ * INNER_, 256), 256, 0, stream>>>(
        wv + (size_t)l * DIM_ * INNER_, wvT + (size_t)l * INNER_ * DIM_, DIM_, INNER_);
    castT_kernel<<<cdiv(INNER_ * DIM_, 256), 256, 0, stream>>>(
        wo + (size_t)l * INNER_ * DIM_, woT + (size_t)l * DIM_ * INNER_, INNER_, DIM_);
    castT_kernel<<<cdiv(DIM_ * FF_, 256), 256, 0, stream>>>(
        ff1_w + (size_t)l * DIM_ * FF_, f1T + (size_t)l * FF_ * DIM_, DIM_, FF_);
    castT_kernel<<<cdiv(FF_ * DIM_, 256), 256, 0, stream>>>(
        ff2_w + (size_t)l * FF_ * DIM_, f2T + (size_t)l * DIM_ * FF_, FF_, DIM_);
    projscale_kernel<<<cdiv(M_ * DH_, 256), 256, 0, stream>>>(
        proj + (size_t)l * M_ * DH_, pjs + (size_t)l * M_ * DH_);
  }

  for (int l = 0; l < DEPTH_; ++l) {
    const f16* wqT_l = wqT + (size_t)l * INNER_ * DIM_;
    const f16* wkT_l = wkT + (size_t)l * INNER_ * DIM_;
    const f16* wvT_l = wvT + (size_t)l * INNER_ * DIM_;
    const f16* woT_l = woT + (size_t)l * DIM_ * INNER_;
    const f16* f1T_l = f1T + (size_t)l * FF_ * DIM_;
    const f16* f2T_l = f2T + (size_t)l * DIM_ * FF_;
    const f16* pjs_l = pjs + (size_t)l * M_ * DH_;

    // --- attention ---
    ln_kernel<<<NT_, DIM_, 0, stream>>>(h, ln1_g + l * DIM_, ln1_b + l * DIM_, y16);
    gemm_kernel<0><<<dim3(INNER_ / 64, NT_ / 256), 256, 0, stream>>>(
        y16, wqT_l, nullptr, q16, NT_, INNER_, DIM_);
    gemm_kernel<0><<<dim3(INNER_ / 64, NT_ / 256), 256, 0, stream>>>(
        y16, wkT_l, nullptr, k16, NT_, INNER_, DIM_);
    gemm_kernel<0><<<dim3(INNER_ / 64, NT_ / 256), 256, 0, stream>>>(
        y16, wvT_l, nullptr, v16, NT_, INNER_, DIM_);
    // FAVOR+ features: dd = q @ (norm*proj)^T viewed as [RQ_,DH] -> [RQ_,M]
    gemm_kernel<0><<<dim3(M_ / 64, RQ_ / 256), 256, 0, stream>>>(
        q16, pjs_l, nullptr, dd16, RQ_, M_, DH_);
    qfeat_kernel<<<RQ_, DIM_, 0, stream>>>(q16, dd16, qp16);
    gemm_kernel<0><<<dim3(M_ / 64, RQ_ / 256), 256, 0, stream>>>(
        k16, pjs_l, nullptr, dd16, RQ_, M_, DH_);
    krow_kernel<<<RQ_, DIM_, 0, stream>>>(k16, dd16, diag, rowmax);
    kreduce_kernel<<<B_ * H_, 256, 0, stream>>>(rowmax, stabk);
    kapply_kernel<<<cdiv(RQ_ * M_, 256), 256, 0, stream>>>(dd16, diag, stabk, kp16);
    // chunked causal linear attention (writes o16 == q16 region)
    attn_kernel<<<B_ * H_, 256, 0, stream>>>(qp16, kp16, v16, o16);
    // h += o @ wo + wo_b
    gemm_kernel<3><<<dim3(DIM_ / 64, NT_ / 256), 256, 0, stream>>>(
        o16, woT_l, wo_b + l * DIM_, h, NT_, DIM_, INNER_);

    // --- FFN ---
    ln_kernel<<<NT_, DIM_, 0, stream>>>(h, ln2_g + l * DIM_, ln2_b + l * DIM_, y16);
    gemm_kernel<2><<<dim3(FF_ / 64, NT_ / 256), 256, 0, stream>>>(
        y16, f1T_l, ff1_b + l * FF_, ff16, NT_, FF_, DIM_);
    gemm_kernel<3><<<dim3(DIM_ / 64, NT_ / 256), 256, 0, stream>>>(
        ff16, f2T_l, ff2_b + l * DIM_, h, NT_, DIM_, FF_);
  }

  pool_kernel<<<cdiv(B_ * DIM_, 256), 256, 0, stream>>>(h, pooled);
  final_kernel<<<1, 64, 0, stream>>>(pooled, fin_w, fin_b, out);
}
